// MultiPathSparseAttention_69449621176440
// MI455X (gfx1250) — matrix operand order repaired and checked by reading, not compiled, not run on hardware
//
#include <hip/hip_runtime.h>
#include <math.h>

// ---------------- problem constants ----------------
#define BB 2
#define LL 2048
#define DD 512
#define NH 8
#define HD 64
#define LCOMP 512          // compressed KV length (ratio 4)
#define NSEL 8             // top-u selected queries
#define SCALE 0.125f       // 1/sqrt(HD)
#define LOG_LL 7.6246189861593985f

typedef __attribute__((ext_vector_type(16))) __bf16 v16bf;
typedef __attribute__((ext_vector_type(8)))  float  v8f;
typedef __attribute__((ext_vector_type(4)))  unsigned int v4u;
typedef __attribute__((ext_vector_type(2)))  unsigned int v2u;
typedef __attribute__((ext_vector_type(4)))  float  v4f;

union Frag {
  v16bf v;
  unsigned short us[16];
  unsigned int   u32[8];
  v4u            q[2];
};

// ---- forced global-address-space accessors (emit global_load/global_store,
// ---- never flat_*: FLAT would tie WMMA waits to DScnt as well as LOADcnt) ----
#define GAS __attribute__((address_space(1)))
__device__ __forceinline__ unsigned int ldg_u32(const unsigned short* p) {
  return *(const GAS unsigned int*)(unsigned long long)p;
}
__device__ __forceinline__ v4u ldg_v4u(const unsigned short* p) {
  return *(const GAS v4u*)(unsigned long long)p;
}
__device__ __forceinline__ unsigned short ldg_u16(const unsigned short* p) {
  return *(const GAS unsigned short*)(unsigned long long)p;
}
__device__ __forceinline__ float ldg_f32(const float* p) {
  return *(const GAS float*)(unsigned long long)p;
}
__device__ __forceinline__ v4f ldg_v4f(const float* p) {
  return *(const GAS v4f*)(unsigned long long)p;
}
__device__ __forceinline__ int ldg_i32(const int* p) {
  return *(const GAS int*)(unsigned long long)p;
}
__device__ __forceinline__ void stg_u16(unsigned short* p, unsigned short v) {
  *(GAS unsigned short*)(unsigned long long)p = v;
}
__device__ __forceinline__ void stg_f32(float* p, float v) {
  *(GAS float*)(unsigned long long)p = v;
}
__device__ __forceinline__ void stg_v2u(unsigned short* p, v2u v) {
  *(GAS v2u*)(unsigned long long)p = v;
}
__device__ __forceinline__ void stg_i32(int* p, int v) {
  *(GAS int*)(unsigned long long)p = v;
}

__device__ __forceinline__ float bf2f(unsigned short h) {
  unsigned int u = ((unsigned int)h) << 16;
  float f; __builtin_memcpy(&f, &u, 4); return f;
}
__device__ __forceinline__ unsigned short f2bf(float f) {
  unsigned int u; __builtin_memcpy(&u, &f, 4);
  u = (u + 0x7FFFu + ((u >> 16) & 1u)) >> 16;   // round-to-nearest-even
  return (unsigned short)u;
}
// pack two floats into a bf16x2 dword (lo -> [15:0], hi -> [31:16])
__device__ __forceinline__ unsigned int pkbf(float lo, float hi) {
#if __has_builtin(__builtin_amdgcn_cvt_pk_bf16_f32)
  auto r = __builtin_amdgcn_cvt_pk_bf16_f32(lo, hi);
  unsigned int u; __builtin_memcpy(&u, &r, 4); return u;
#else
  return (unsigned int)f2bf(lo) | ((unsigned int)f2bf(hi) << 16);
#endif
}
__device__ __forceinline__ v8f wmma_bf16(const Frag& a, const Frag& b, v8f c) {
  return __builtin_amdgcn_wmma_f32_16x16x32_bf16(false, a.v, false, b.v,
                                                 (short)0, c, false, false);
}
__device__ __forceinline__ void path_w(const float* mix, float pw[3]) {
  float m0 = ldg_f32(mix), m1 = ldg_f32(mix + 1), m2 = ldg_f32(mix + 2);
  float mx = fmaxf(m0, fmaxf(m1, m2));
  float e0 = __expf(m0 - mx), e1 = __expf(m1 - mx), e2 = __expf(m2 - mx);
  float inv = 1.f / (e0 + e1 + e2);
  pw[0] = e0 * inv; pw[1] = e1 * inv; pw[2] = e2 * inv;
}

// ============================================================================
// Flash tile: one wave handles a 16-query tile against Lk keys (chunks of 32).
// Computes S^T = K * Q^T so the S^T C-tile column == query: online softmax is
// per-lane scalar math + one shfl_xor(16), and the P A-fragment is a pure
// register reinterpret of the S^T C-tiles (C vgpr e -> A elem e / 8+e).
// BAND: sliding-window mask (|j-i| <= 64) for output + full-row lse/mean stats.
// ============================================================================
template<bool BAND>
__device__ void flash_tile(const unsigned short* __restrict__ Kp, int kstride,
                           const unsigned short* __restrict__ Vt, int Lk,
                           const Frag& qb0, const Frag& qb1, int q0,
                           v8f* acc, float& M, float& Z,
                           float& Mf, float& Zf, float& Sf)
{
  const int lane = threadIdx.x & 31;
  const int half = lane >> 4;
  const int nq   = lane & 15;
  const int iq   = q0 + nq;

  for (int c = 0; c < Lk; c += 32) {
    bool doOut = true;
    if (BAND) doOut = (c + 31 >= q0 - 64) && (c <= q0 + 15 + 64);

    // prefetch next chunk's K rows (global_prefetch_b8)
    if (c + 32 < Lk) {
      __builtin_prefetch(Kp + (size_t)(c + 32 + nq) * kstride, 0, 1);
      __builtin_prefetch(Kp + (size_t)(c + 48 + nq) * kstride, 0, 1);
    }

    // ---- S^T tiles: s0 = keys c..c+15, s1 = keys c+16..c+31 ----
    v8f s0 = {0.f,0.f,0.f,0.f,0.f,0.f,0.f,0.f};
    v8f s1 = s0;
    {
      const unsigned short* kr0 = Kp + (size_t)(c + nq) * kstride;
      const unsigned short* kr1 = Kp + (size_t)(c + 16 + nq) * kstride;
      Frag a;
#pragma unroll
      for (int r = 0; r < 4; ++r) {
        a.u32[r]     = ldg_u32(kr0 + 2*r + 16*half);
        a.u32[r + 4] = ldg_u32(kr0 + 8 + 2*r + 16*half);
      }
      s0 = wmma_bf16(a, qb0, s0);
#pragma unroll
      for (int r = 0; r < 4; ++r) {
        a.u32[r]     = ldg_u32(kr0 + 32 + 2*r + 16*half);
        a.u32[r + 4] = ldg_u32(kr0 + 40 + 2*r + 16*half);
      }
      s0 = wmma_bf16(a, qb1, s0);
#pragma unroll
      for (int r = 0; r < 4; ++r) {
        a.u32[r]     = ldg_u32(kr1 + 2*r + 16*half);
        a.u32[r + 4] = ldg_u32(kr1 + 8 + 2*r + 16*half);
      }
      s1 = wmma_bf16(a, qb0, s1);
#pragma unroll
      for (int r = 0; r < 4; ++r) {
        a.u32[r]     = ldg_u32(kr1 + 32 + 2*r + 16*half);
        a.u32[r + 4] = ldg_u32(kr1 + 40 + 2*r + 16*half);
      }
      s1 = wmma_bf16(a, qb1, s1);
    }

    float se0[8], se1[8];
#pragma unroll
    for (int r = 0; r < 8; ++r) { se0[r] = s0[r]; se1[r] = s1[r]; }

    if (BAND) {
      // --- full-row stats (unmasked): online logsumexp + running sum ---
      float cmax = -1e30f, csum = 0.f;
#pragma unroll
      for (int r = 0; r < 8; ++r) {
        cmax = fmaxf(cmax, fmaxf(se0[r], se1[r]));
        csum += se0[r] + se1[r];
      }
      cmax = fmaxf(cmax, __shfl_xor(cmax, 16));
      float nM = fmaxf(Mf, cmax);
      float es = 0.f;
#pragma unroll
      for (int r = 0; r < 8; ++r)
        es += __expf(se0[r] - nM) + __expf(se1[r] - nM);
      es += __shfl_xor(es, 16);
      Zf = Zf * __expf(Mf - nM) + es;
      Mf = nM;
      Sf += csum + __shfl_xor(csum, 16);
      // --- band mask for the local output path ---
      if (doOut) {
#pragma unroll
        for (int r = 0; r < 8; ++r) {
          int j0 = c + r + 8*half;
          int j1 = j0 + 16;
          if (j0 < iq - 64 || j0 > iq + 64) se0[r] = -1e30f;
          if (j1 < iq - 64 || j1 > iq + 64) se1[r] = -1e30f;
        }
      }
    }

    if (doOut) {
      if (c + 32 < Lk)   // prefetch next V chunk rows used by this wave
        __builtin_prefetch(Vt + (size_t)nq * Lk + c + 32, 0, 1);

      float cmax = -1e30f;
#pragma unroll
      for (int r = 0; r < 8; ++r) cmax = fmaxf(cmax, fmaxf(se0[r], se1[r]));
      cmax = fmaxf(cmax, __shfl_xor(cmax, 16));
      float nM = fmaxf(M, cmax);
      float sc_old = __expf(M - nM);

      // P A-fragment from S^T C-tiles (registers only); packed bf16x2 converts
      Frag pa;
      float p0[8], p1[8], zc = 0.f;
#pragma unroll
      for (int r = 0; r < 8; ++r) {
        p0[r] = __expf(se0[r] - nM);
        p1[r] = __expf(se1[r] - nM);
        zc += p0[r] + p1[r];
      }
#pragma unroll
      for (int j = 0; j < 4; ++j) {
        pa.u32[j]     = pkbf(p0[2*j], p0[2*j + 1]);
        pa.u32[j + 4] = pkbf(p1[2*j], p1[2*j + 1]);
      }
      zc += __shfl_xor(zc, 16);
      Z = Z * sc_old + zc;
      M = nM;

      // rescale O accumulators: per-row factor broadcast from stat lanes
      float fac[8];
#pragma unroll
      for (int r = 0; r < 8; ++r) fac[r] = __shfl(sc_old, r + 8*half);
#pragma unroll
      for (int nt = 0; nt < 4; ++nt)
#pragma unroll
        for (int r = 0; r < 8; ++r) acc[nt][r] *= fac[r];

      // O += P @ V : V B-fragments are two 16B per-lane loads from Vt (keys 8h.., 16+8h..)
#pragma unroll
      for (int nt = 0; nt < 4; ++nt) {
        const unsigned short* vr =
            Vt + (size_t)(nt*16 + nq) * Lk + c + 8*half;
        Frag vb;
        vb.q[0] = ldg_v4u(vr);
        vb.q[1] = ldg_v4u(vr + 16);
        acc[nt] = wmma_bf16(pa, vb, acc[nt]);
      }
    }
  }
}

// ============================================================================
// bf16 WMMA GEMM body: C(MxN) = A(MxK) @ W(KxN) + bias, W pre-transposed.
// One wave computes a 32x64 block (2 m-tiles x 4 n-tiles): every W B-fragment
// feeds two WMMAs, halving Wt traffic. ACT: 0=none, 1=exact GELU.
// ============================================================================
template<int ACT, int OUT_BF16>
__device__ void gemm_body(const unsigned short* __restrict__ A,
                          const unsigned short* __restrict__ Wt,
                          const float* __restrict__ bias,
                          void* __restrict__ Cout,
                          int Mrows, int N, int K, int wid)
{
  int ntiles = N >> 6;
  int mt = wid / ntiles, nt0 = wid % ntiles;
  if (mt * 32 >= Mrows) return;
  int lane = threadIdx.x & 31, half = lane >> 4, ln = lane & 15;
  int m0 = mt * 32, n0 = nt0 * 64;

  v8f z = {0.f,0.f,0.f,0.f,0.f,0.f,0.f,0.f};
  v8f acc[2][4] = { { z, z, z, z }, { z, z, z, z } };
  const unsigned short* arow0 = A + (size_t)(m0 + ln) * K;
  const unsigned short* arow1 = arow0 + (size_t)16 * K;

  for (int kc = 0; kc < K; kc += 32) {
    Frag a0, a1;
#pragma unroll
    for (int r = 0; r < 4; ++r) {
      a0.u32[r]     = ldg_u32(arow0 + kc + 2*r + 8*half);
      a0.u32[r + 4] = ldg_u32(arow0 + kc + 16 + 2*r + 8*half);
      a1.u32[r]     = ldg_u32(arow1 + kc + 2*r + 8*half);
      a1.u32[r + 4] = ldg_u32(arow1 + kc + 16 + 2*r + 8*half);
    }
#pragma unroll
    for (int nt = 0; nt < 4; ++nt) {
      const unsigned short* wr = Wt + (size_t)(n0 + nt*16 + ln) * K + kc + 8*half;
      Frag b;
      b.q[0] = ldg_v4u(wr);
      b.q[1] = ldg_v4u(wr + 16);
      acc[0][nt] = wmma_bf16(a0, b, acc[0][nt]);
      acc[1][nt] = wmma_bf16(a1, b, acc[1][nt]);
    }
  }
#pragma unroll
  for (int mi = 0; mi < 2; ++mi)
#pragma unroll
    for (int nt = 0; nt < 4; ++nt) {
      float bv = bias ? ldg_f32(bias + n0 + nt*16 + ln) : 0.f;
#pragma unroll
      for (int r = 0; r < 8; ++r) {
        float x = acc[mi][nt][r] + bv;
        if (ACT == 1) x = 0.5f * x * (1.f + erff(x * 0.70710678118654752f));
        int m = m0 + mi*16 + r + 8*half;
        size_t off = (size_t)m * N + n0 + nt*16 + ln;
        if (OUT_BF16) stg_u16((unsigned short*)Cout + off, f2bf(x));
        else          stg_f32((float*)Cout + off, x);
      }
    }
}

template<int ACT, int OUT_BF16>
__global__ __launch_bounds__(256)
void k_gemm(const unsigned short* __restrict__ A,
            const unsigned short* __restrict__ Wt,
            const float* __restrict__ bias,
            void* __restrict__ Cout, int Mrows, int N, int K)
{
  int wid = (blockIdx.x * blockDim.x + threadIdx.x) >> 5;
  gemm_body<ACT, OUT_BF16>(A, Wt, bias, Cout, Mrows, N, K, wid);
}

struct GemmSet { const unsigned short* A; const unsigned short* Wt;
                 const float* bias; unsigned short* C; };
__global__ __launch_bounds__(256)
void k_gemm_qkv(GemmSet g0, GemmSet g1, GemmSet g2, int Mrows, int N, int K)
{
  GemmSet g = (blockIdx.y == 0) ? g0 : (blockIdx.y == 1 ? g1 : g2);
  int wid = (blockIdx.x * blockDim.x + threadIdx.x) >> 5;
  gemm_body<0, 1>(g.A, g.Wt, g.bias, g.C, Mrows, N, K, wid);
}

// ---------------- fused prep kernels ----------------
struct Cast3Args { const float* src[3]; unsigned short* dst[3]; int n; };
__global__ __launch_bounds__(256)
void k_cast3(Cast3Args a) {                      // 4 elems/thread, bf16x2 packs
  int i = (blockIdx.x * blockDim.x + threadIdx.x) * 4;
  if (i < a.n) {
    const v4f f = ldg_v4f(a.src[blockIdx.y] + i);
    v2u o; o.x = pkbf(f.x, f.y); o.y = pkbf(f.z, f.w);
    stg_v2u(a.dst[blockIdx.y] + i, o);
  }
}
__global__ __launch_bounds__(256)
void k_cast1(const float* __restrict__ src,
             unsigned short* __restrict__ dst, int n) {
  int i = (blockIdx.x * blockDim.x + threadIdx.x) * 4;
  if (i < n) {
    const v4f f = ldg_v4f(src + i);
    v2u o; o.x = pkbf(f.x, f.y); o.y = pkbf(f.z, f.w);
    stg_v2u(dst + i, o);
  }
}
// Wt[n*K + k] = W[k*N + n] for 6 weight matrices (512x512)
struct TC6Args { const float* W[6]; unsigned short* Wt[6]; };
__global__ __launch_bounds__(256)
void k_tc6(TC6Args a) {
  int i = blockIdx.x * blockDim.x + threadIdx.x;
  if (i < DD * DD) {
    int n = i >> 9, k = i & 511;
    stg_u16(a.Wt[blockIdx.y] + i, f2bf(ldg_f32(a.W[blockIdx.y] + (size_t)k * DD + n)));
  }
}
// mean-pool groups of 4 rows for both K and V projections
__global__ __launch_bounds__(256)
void k_pool4(const unsigned short* __restrict__ kp,
             const unsigned short* __restrict__ vp,
             unsigned short* __restrict__ kcr,
             unsigned short* __restrict__ vc, int rows) {
  int i = blockIdx.x * blockDim.x + threadIdx.x;
  if (i < rows * DD) {
    int d = i % DD;
    size_t base = (size_t)(i / DD) * 4 * DD + d;
    float a = 0.25f * (bf2f(ldg_u16(kp + base)) + bf2f(ldg_u16(kp + base + DD)) +
                       bf2f(ldg_u16(kp + base + 2*DD)) + bf2f(ldg_u16(kp + base + 3*DD)));
    float b = 0.25f * (bf2f(ldg_u16(vp + base)) + bf2f(ldg_u16(vp + base + DD)) +
                       bf2f(ldg_u16(vp + base + 2*DD)) + bf2f(ldg_u16(vp + base + 3*DD)));
    stg_u16(kcr + i, f2bf(a));
    stg_u16(vc + i, f2bf(b));
  }
}
// (B, Lr, D) -> (B, H, HD, Lr); y=0: full V (LL), y=1: compressed V (LCOMP)
__global__ __launch_bounds__(256)
void k_thead2(const unsigned short* __restrict__ v0,
              unsigned short* __restrict__ t0,
              const unsigned short* __restrict__ v1,
              unsigned short* __restrict__ t1) {
  int Lr = blockIdx.y ? LCOMP : LL;
  const unsigned short* src = blockIdx.y ? v1 : v0;
  unsigned short* dst = blockIdx.y ? t1 : t0;
  int i = blockIdx.x * blockDim.x + threadIdx.x;
  if (i < BB * Lr * DD) {
    int d = i % DD;
    int l = (i / DD) % Lr;
    int b = i / (DD * Lr);
    stg_u16(dst + (((size_t)(b * NH + d / HD)) * HD + d % HD) * Lr + l,
            ldg_u16(src + i));
  }
}

// ---------------- attention path kernels ----------------
__global__ __launch_bounds__(256)
void k_global_attn(const unsigned short* __restrict__ qproj,
                   const unsigned short* __restrict__ kc,
                   const unsigned short* __restrict__ vct,
                   const float* __restrict__ mixer,
                   float* __restrict__ combined)
{
  int wid = (blockIdx.x * blockDim.x + threadIdx.x) >> 5;
  int qt = wid & 127, bh = wid >> 7;
  int h = bh & 7, b = bh >> 3;
  int lane = threadIdx.x & 31, half = lane >> 4, nq = lane & 15;
  int q0 = qt * 16;
  float pw[3]; path_w(mixer, pw);

  const unsigned short* qrow = qproj + ((size_t)(b * LL + q0 + nq)) * DD + h * HD;
  Frag qb0, qb1;
#pragma unroll
  for (int j = 0; j < 8; ++j) {
    qb0.u32[j] = pkbf(bf2f(ldg_u16(qrow + 2*j + 16*half)) * SCALE,
                      bf2f(ldg_u16(qrow + 2*j + 1 + 16*half)) * SCALE);
    qb1.u32[j] = pkbf(bf2f(ldg_u16(qrow + 32 + 2*j + 16*half)) * SCALE,
                      bf2f(ldg_u16(qrow + 33 + 2*j + 16*half)) * SCALE);
  }
  v8f z = {0.f,0.f,0.f,0.f,0.f,0.f,0.f,0.f};
  v8f acc[4] = { z, z, z, z };
  float M = -1e30f, Z = 0.f, dM = -1e30f, dZ = 0.f, dS = 0.f;
  flash_tile<false>(kc + (size_t)b * LCOMP * DD + h * HD, DD,
                    vct + ((size_t)(b * NH + h)) * HD * LCOMP, LCOMP,
                    qb0, qb1, q0, acc, M, Z, dM, dZ, dS);
  float zinv = 1.f / Z;
  float zr[8];
#pragma unroll
  for (int r = 0; r < 8; ++r) zr[r] = __shfl(zinv, r + 8*half);
#pragma unroll
  for (int nt = 0; nt < 4; ++nt)
#pragma unroll
    for (int r = 0; r < 8; ++r)
      stg_f32(combined +
              ((size_t)(b * LL + q0 + r + 8*half)) * DD + h * HD + nt*16 + nq,
              pw[0] * acc[nt][r] * zr[r]);
}

__global__ __launch_bounds__(256)
void k_local_attn(const unsigned short* __restrict__ qproj,
                  const unsigned short* __restrict__ kproj,
                  const unsigned short* __restrict__ vt,
                  const float* __restrict__ mixer,
                  float* __restrict__ combined,
                  float* __restrict__ importance)
{
  int wid = (blockIdx.x * blockDim.x + threadIdx.x) >> 5;
  int qt = wid & 127, bh = wid >> 7;
  int h = bh & 7, b = bh >> 3;
  int lane = threadIdx.x & 31, half = lane >> 4, nq = lane & 15;
  int q0 = qt * 16;
  float pw[3]; path_w(mixer, pw);

  const unsigned short* qrow = qproj + ((size_t)(b * LL + q0 + nq)) * DD + h * HD;
  Frag qb0, qb1;
#pragma unroll
  for (int j = 0; j < 8; ++j) {
    qb0.u32[j] = pkbf(bf2f(ldg_u16(qrow + 2*j + 16*half)) * SCALE,
                      bf2f(ldg_u16(qrow + 2*j + 1 + 16*half)) * SCALE);
    qb1.u32[j] = pkbf(bf2f(ldg_u16(qrow + 32 + 2*j + 16*half)) * SCALE,
                      bf2f(ldg_u16(qrow + 33 + 2*j + 16*half)) * SCALE);
  }
  v8f z = {0.f,0.f,0.f,0.f,0.f,0.f,0.f,0.f};
  v8f acc[4] = { z, z, z, z };
  float M = -1e30f, Z = 0.f, Mf = -1e30f, Zf = 0.f, Sf = 0.f;
  flash_tile<true>(kproj + (size_t)b * LL * DD + h * HD, DD,
                   vt + ((size_t)(b * NH + h)) * HD * LL, LL,
                   qb0, qb1, q0, acc, M, Z, Mf, Zf, Sf);
  float zinv = 1.f / Z;
  float zr[8];
#pragma unroll
  for (int r = 0; r < 8; ++r) zr[r] = __shfl(zinv, r + 8*half);
#pragma unroll
  for (int nt = 0; nt < 4; ++nt)
#pragma unroll
    for (int r = 0; r < 8; ++r) {
      float* p = combined +
          ((size_t)(b * LL + q0 + r + 8*half)) * DD + h * HD + nt*16 + nq;
      stg_f32(p, ldg_f32(p) + pw[1] * acc[nt][r] * zr[r]);
    }
  if (lane < 16)
    stg_f32(importance + (size_t)bh * LL + q0 + nq,
            Mf + __logf(Zf) - LOG_LL - Sf * (1.f / LL));
}

// top-8 argmax per (b,h)
__global__ __launch_bounds__(256)
void k_top8(const float* __restrict__ imp, int* __restrict__ idxb) {
  __shared__ float simp[LL];
  __shared__ float rmax[8];
  __shared__ int   ridx[8];
  int bh = blockIdx.x, t = threadIdx.x;
  for (int i = t; i < LL; i += 256) simp[i] = ldg_f32(imp + (size_t)bh * LL + i);
  __syncthreads();
  for (int u = 0; u < NSEL; ++u) {
    float best = -1e30f; int bi = 0;
    for (int i = t; i < LL; i += 256)
      if (simp[i] > best) { best = simp[i]; bi = i; }
#pragma unroll
    for (int off = 16; off > 0; off >>= 1) {
      float ob = __shfl_down(best, off);
      int   oi = __shfl_down(bi, off);
      if (ob > best) { best = ob; bi = oi; }
    }
    if ((t & 31) == 0) { rmax[t >> 5] = best; ridx[t >> 5] = bi; }
    __syncthreads();
    if (t == 0) {
      for (int w = 1; w < 8; ++w)
        if (rmax[w] > best) { best = rmax[w]; bi = ridx[w]; }
      stg_i32(idxb + bh * NSEL + u, bi);
      simp[bi] = -1e30f;
    }
    __syncthreads();
  }
}

// selection path: one wave per (b,h); 8 selected queries padded to 16 slots
__global__ __launch_bounds__(32)
void k_sel_attn(const unsigned short* __restrict__ qproj,
                const unsigned short* __restrict__ kproj,
                const unsigned short* __restrict__ vt,
                const int* __restrict__ idxb,
                const float* __restrict__ mixer,
                float* __restrict__ combined)
{
  int bh = blockIdx.x;
  int h = bh & 7, b = bh >> 3;
  int lane = threadIdx.x & 31, half = lane >> 4, nq = lane & 15;
  float pw[3]; path_w(mixer, pw);

  int qrow_i = ldg_i32(idxb + bh * NSEL + (nq & 7));
  const unsigned short* qrow = qproj + ((size_t)(b * LL + qrow_i)) * DD + h * HD;
  Frag qb0, qb1;
#pragma unroll
  for (int j = 0; j < 8; ++j) {
    qb0.u32[j] = pkbf(bf2f(ldg_u16(qrow + 2*j + 16*half)) * SCALE,
                      bf2f(ldg_u16(qrow + 2*j + 1 + 16*half)) * SCALE);
    qb1.u32[j] = pkbf(bf2f(ldg_u16(qrow + 32 + 2*j + 16*half)) * SCALE,
                      bf2f(ldg_u16(qrow + 33 + 2*j + 16*half)) * SCALE);
  }
  v8f z = {0.f,0.f,0.f,0.f,0.f,0.f,0.f,0.f};
  v8f acc[4] = { z, z, z, z };
  float M = -1e30f, Z = 0.f, dM = -1e30f, dZ = 0.f, dS = 0.f;
  flash_tile<false>(kproj + (size_t)b * LL * DD + h * HD, DD,
                    vt + ((size_t)(b * NH + h)) * HD * LL, LL,
                    qb0, qb1, 0, acc, M, Z, dM, dZ, dS);
  float zinv = 1.f / Z;
  float zr[8];
#pragma unroll
  for (int r = 0; r < 8; ++r) zr[r] = __shfl(zinv, r + 8*half);
  if (half == 0) {   // rows m = 0..7 are the real (distinct) selected queries
#pragma unroll
    for (int r = 0; r < 8; ++r) {
      int qr = ldg_i32(idxb + bh * NSEL + r);
#pragma unroll
      for (int nt = 0; nt < 4; ++nt) {
        float* p = combined + ((size_t)(b * LL + qr)) * DD + h * HD + nt*16 + nq;
        stg_f32(p, ldg_f32(p) + pw[2] * acc[nt][r] * zr[r]);
      }
    }
  }
}

// ============================================================================
extern "C" void kernel_launch(void* const* d_in, const int* in_sizes, int n_in,
                              void* d_out, int out_size, void* d_ws, size_t ws_size,
                              hipStream_t stream) {
  const float* query = (const float*)d_in[0];
  const float* key   = (const float*)d_in[1];
  const float* value = (const float*)d_in[2];
  const float* Wq = (const float*)d_in[3];  const float* bq = (const float*)d_in[4];
  const float* Wk = (const float*)d_in[5];  const float* bk = (const float*)d_in[6];
  const float* Wv = (const float*)d_in[7];  const float* bv = (const float*)d_in[8];
  const float* Wo = (const float*)d_in[9];  const float* bo = (const float*)d_in[10];
  const float* Wc1 = (const float*)d_in[11]; const float* bc1 = (const float*)d_in[12];
  const float* Wc2 = (const float*)d_in[13]; const float* bc2 = (const float*)d_in[14];
  const float* mixer = (const float*)d_in[15];

  // ---- workspace carve-up ----
  size_t off = 0;
  auto carve = [&](size_t bytes) {
    void* p = (char*)d_ws + off;
    off += (bytes + 255) & ~(size_t)255;
    return p;
  };
  const size_t NTOK = (size_t)BB * LL;          // 4096
  const size_t NCMP = (size_t)BB * LCOMP;       // 1024
  unsigned short* qb    = (unsigned short*)carve(NTOK * DD * 2);
  unsigned short* kb    = (unsigned short*)carve(NTOK * DD * 2);
  unsigned short* vb    = (unsigned short*)carve(NTOK * DD * 2);
  unsigned short* WqT   = (unsigned short*)carve((size_t)DD * DD * 2);
  unsigned short* WkT   = (unsigned short*)carve((size_t)DD * DD * 2);
  unsigned short* WvT   = (unsigned short*)carve((size_t)DD * DD * 2);
  unsigned short* WoT   = (unsigned short*)carve((size_t)DD * DD * 2);
  unsigned short* Wc1T  = (unsigned short*)carve((size_t)DD * DD * 2);
  unsigned short* Wc2T  = (unsigned short*)carve((size_t)DD * DD * 2);
  unsigned short* qproj = (unsigned short*)carve(NTOK * DD * 2);
  unsigned short* kproj = (unsigned short*)carve(NTOK * DD * 2);
  unsigned short* vproj = (unsigned short*)carve(NTOK * DD * 2);
  unsigned short* kcr   = (unsigned short*)carve(NCMP * DD * 2);
  unsigned short* vc    = (unsigned short*)carve(NCMP * DD * 2);
  unsigned short* hid   = (unsigned short*)carve(NCMP * DD * 2);
  unsigned short* kc    = (unsigned short*)carve(NCMP * DD * 2);
  unsigned short* vt    = (unsigned short*)carve(NTOK * DD * 2);
  unsigned short* vct   = (unsigned short*)carve(NCMP * DD * 2);
  float* combined       = (float*)carve(NTOK * DD * 4);
  unsigned short* combb = (unsigned short*)carve(NTOK * DD * 2);
  float* importance     = (float*)carve((size_t)BB * NH * LL * 4);
  int*   idxb           = (int*)carve((size_t)BB * NH * NSEL * 4);
  (void)ws_size; (void)n_in; (void)in_sizes; (void)out_size;

  const int T = 256;
  auto blks = [](size_t n, int t) { return (int)((n + t - 1) / t); };
  const size_t NE = NTOK * DD;                  // 2M elements

  // 1) cast q/k/v inputs to bf16 (fused, vectorized x4)
  {
    Cast3Args ca; ca.src[0] = query; ca.src[1] = key; ca.src[2] = value;
    ca.dst[0] = qb; ca.dst[1] = kb; ca.dst[2] = vb; ca.n = (int)NE;
    dim3 g(blks(NE / 4, T), 3);
    k_cast3<<<g, T, 0, stream>>>(ca);
  }
  // 2) transpose+cast all six weight matrices (fused)
  {
    TC6Args ta;
    ta.W[0] = Wq;  ta.Wt[0] = WqT;  ta.W[1] = Wk;  ta.Wt[1] = WkT;
    ta.W[2] = Wv;  ta.Wt[2] = WvT;  ta.W[3] = Wo;  ta.Wt[3] = WoT;
    ta.W[4] = Wc1; ta.Wt[4] = Wc1T; ta.W[5] = Wc2; ta.Wt[5] = Wc2T;
    dim3 g(blks((size_t)DD * DD, T), 6);
    k_tc6<<<g, T, 0, stream>>>(ta);
  }
  // 3) q/k/v projections in one launch (gridDim.y selects operand set)
  {
    int waves = (int)(NTOK / 32) * (DD / 64);   // 1024
    dim3 g((waves + 7) / 8, 3);
    GemmSet g0{qb, WqT, bq, qproj}, g1{kb, WkT, bk, kproj}, g2{vb, WvT, bv, vproj};
    k_gemm_qkv<<<g, T, 0, stream>>>(g0, g1, g2, (int)NTOK, DD, DD);
  }
  // 4) mean-pool ratio 4
  k_pool4<<<blks(NCMP * DD, T), T, 0, stream>>>(kproj, vproj, kcr, vc, (int)NCMP);
  // 5) compressor MLP (GELU exact)
  {
    int waves = (int)(NCMP / 32) * (DD / 64);   // 256
    int g = (waves + 7) / 8;
    k_gemm<1,1><<<g, T, 0, stream>>>(kcr, Wc1T, bc1, hid, (int)NCMP, DD, DD);
    k_gemm<0,1><<<g, T, 0, stream>>>(hid, Wc2T, bc2, kc,  (int)NCMP, DD, DD);
  }
  // 6) V transposes (fused: full + compressed)
  k_thead2<<<dim3(blks(NE, T), 2), T, 0, stream>>>(vproj, vt, vc, vct);
  // 7) global path (initializes combined = pw0 * out_g)
  k_global_attn<<<256, T, 0, stream>>>(qproj, kc, vct, mixer, combined);
  // 8) local path (+= pw1 * out_l) and importance stats
  k_local_attn<<<256, T, 0, stream>>>(qproj, kproj, vt, mixer, combined, importance);
  // 9) top-8 selection
  k_top8<<<BB * NH, T, 0, stream>>>(importance, idxb);
  // 10) selection path (+= pw2 at selected rows)
  k_sel_attn<<<BB * NH, 32, 0, stream>>>(qproj, kproj, vt, idxb, mixer, combined);
  // 11) output projection -> d_out (fp32)
  k_cast1<<<blks(NE / 4, T), T, 0, stream>>>(combined, combb, (int)NE);
  {
    int waves = (int)(NTOK / 32) * (DD / 64);
    k_gemm<0,0><<<(waves + 7) / 8, T, 0, stream>>>(combb, WoT, bo, d_out,
                                                   (int)NTOK, DD, DD);
  }
}
